// MinkowskiAttention_69578470195863
// MI455X (gfx1250) — matrix operand order repaired and checked by reading, not compiled, not run on hardware
//
#include <hip/hip_runtime.h>
#include <hip/hip_bf16.h>

// ---------------------------------------------------------------------------
// MinkowskiAttention for MI455X (gfx1250): bf16 WMMA 16x16x32 everywhere,
// async global->LDS staging (ASYNCcnt) for the attention-score tiles.
// ---------------------------------------------------------------------------

typedef __bf16 bf16_t;
typedef __attribute__((ext_vector_type(16))) __bf16 v16bf;
typedef __attribute__((ext_vector_type(8)))  float  v8f;

union Frag16 { v16bf v; unsigned int u[8]; };

static constexpr int kB  = 2;
static constexpr int kT  = 2048;
static constexpr int kD  = 1024;
static constexpr int kH  = 16;
static constexpr int kHD = 64;

#if __has_builtin(__builtin_amdgcn_global_load_async_to_lds_b128)
#define HAVE_ASYNC_LDS 1
#else
#define HAVE_ASYNC_LDS 0
#endif

#define AS_GLOBAL __attribute__((address_space(1)))
#define AS_LDS    __attribute__((address_space(3)))

typedef int v4i_t __attribute__((vector_size(16)));   // matches builtin param

__device__ __forceinline__ unsigned int pack2bf(float a, float b) {
    unsigned short lo = __builtin_bit_cast(unsigned short, (__bf16)a);
    unsigned short hi = __builtin_bit_cast(unsigned short, (__bf16)b);
    return ((unsigned int)hi << 16) | (unsigned int)lo;
}

__device__ __forceinline__ v8f wmma_bf16(const Frag16& a, const Frag16& b, v8f c) {
    return __builtin_amdgcn_wmma_f32_16x16x32_bf16(false, a.v, false, b.v,
                                                   (short)0, c, false, false);
}

// 16-byte global -> LDS copy; async (ASYNCcnt) when the toolchain exposes it.
__device__ __forceinline__ void copy16_g2l(const bf16_t* __restrict__ g,
                                           bf16_t* __restrict__ l) {
#if HAVE_ASYNC_LDS
    __builtin_amdgcn_global_load_async_to_lds_b128(
        (AS_GLOBAL v4i_t*)g, (AS_LDS v4i_t*)l, 0, 0);
#else
    *(uint4*)l = *(const uint4*)g;
#endif
}

__device__ __forceinline__ void wait_async_copies() {
#if HAVE_ASYNC_LDS
#if __has_builtin(__builtin_amdgcn_s_wait_asynccnt)
    __builtin_amdgcn_s_wait_asynccnt(0);
#else
    asm volatile("s_wait_asynccnt 0x0" ::: "memory");
#endif
#endif
}

// ---------------------------------------------------------------------------
// GEMM: C[M,N] = A[M,K] @ W[K,N] + bias   (A,W f32 in memory; bf16 WMMA core)
// WG = 256 threads = 8 waves; WG tile 64x64; wave -> 2 adjacent 16x16 tiles.
// LDS rows padded to 34 bf16 (17 dwords, gcd(17,64)=1 -> conflict-free).
// ---------------------------------------------------------------------------
__global__ __launch_bounds__(256)
void mk_gemm_bias(const float* __restrict__ A, const float* __restrict__ W,
                  const float* __restrict__ bias,
                  float* __restrict__ Cf, bf16_t* __restrict__ Cb,
                  int M, int K, int N)
{
    __shared__ bf16_t As[64 * 34];   // [m][k]
    __shared__ bf16_t Bs[64 * 34];   // [n][k]  (transposed W tile)

    const int tid  = threadIdx.x;
    const int lane = tid & 31;
    const int wave = tid >> 5;
    const int m0   = blockIdx.y * 64;
    const int n0   = blockIdx.x * 64;
    const int tm   = wave >> 1;            // 0..3 : 16-row block
    const int tn0  = (wave & 1) * 2;       // 0 or 2 : first 16-col block
    const int mRow = lane & 15;
    const int half = lane >> 4;

    const int aRow = tid >> 2;             // 0..63
    const int aCol = (tid & 3) * 8;        // 0,8,16,24
    const int wRow = tid >> 3;             // 0..31
    const int wCol = (tid & 7) * 8;        // 0..56

    v8f acc0 = {}; v8f acc1 = {};

    for (int k0 = 0; k0 < K; k0 += 32) {
        // --- stage A tile 64x32 (f32 -> bf16) ---
        const float* ap = A + (size_t)(m0 + aRow) * K + k0 + aCol;
        float4 a0 = *(const float4*)ap;
        float4 a1 = *(const float4*)(ap + 4);
        unsigned int* ad = (unsigned int*)&As[aRow * 34 + aCol];
        ad[0] = pack2bf(a0.x, a0.y); ad[1] = pack2bf(a0.z, a0.w);
        ad[2] = pack2bf(a1.x, a1.y); ad[3] = pack2bf(a1.z, a1.w);
        if (k0 + 32 < K) __builtin_prefetch(ap + 32, 0, 1);   // global_prefetch_b8

        // --- stage W tile 32x64 transposed (f32 -> bf16) ---
        const float* wp = W + (size_t)(k0 + wRow) * N + n0 + wCol;
        float4 w0 = *(const float4*)wp;
        float4 w1 = *(const float4*)(wp + 4);
        float wv[8] = {w0.x, w0.y, w0.z, w0.w, w1.x, w1.y, w1.z, w1.w};
        #pragma unroll
        for (int j = 0; j < 8; ++j)
            Bs[(wCol + j) * 34 + wRow] = (bf16_t)wv[j];
        __syncthreads();

        // --- fragments from LDS ---
        Frag16 af;
        {
            const unsigned int* ar = (const unsigned int*)&As[(tm * 16 + mRow) * 34];
            #pragma unroll
            for (int v = 0; v < 4; ++v) af.u[v]     = ar[v + 8 * half];
            #pragma unroll
            for (int v = 0; v < 4; ++v) af.u[4 + v] = ar[4 + v + 8 * half];
        }
        Frag16 bf0, bf1;
        {
            const unsigned int* b0 = (const unsigned int*)&Bs[(tn0 * 16 + mRow) * 34 + 16 * half];
            const unsigned int* b1 = (const unsigned int*)&Bs[((tn0 + 1) * 16 + mRow) * 34 + 16 * half];
            #pragma unroll
            for (int v = 0; v < 8; ++v) { bf0.u[v] = b0[v]; bf1.u[v] = b1[v]; }
        }
        acc0 = wmma_bf16(af, bf0, acc0);
        acc1 = wmma_bf16(af, bf1, acc1);
        __syncthreads();
    }

    // --- epilogue: C layout lane n = L%16, VGPR r -> row r + 8*half ---
    const int cn0 = n0 + tn0 * 16 + mRow;
    const int cn1 = cn0 + 16;
    const float bv0 = bias ? bias[cn0] : 0.0f;
    const float bv1 = bias ? bias[cn1] : 0.0f;
    #pragma unroll
    for (int r = 0; r < 8; ++r) {
        const int cm = m0 + tm * 16 + r + 8 * half;
        const float o0 = acc0[r] + bv0;
        const float o1 = acc1[r] + bv1;
        if (Cf) {
            Cf[(size_t)cm * N + cn0] = o0;
            Cf[(size_t)cm * N + cn1] = o1;
        } else {
            Cb[(size_t)cm * N + cn0] = (bf16_t)o0;
            Cb[(size_t)cm * N + cn1] = (bf16_t)o1;
        }
    }
}

// ---------------------------------------------------------------------------
// Scores: S[b,h,i,j] = scale * (Q_bh K_bh^T)[i,j] - relu(ds2[b,i,j])
// Q tile (128x64 bf16) and K tile (16x64 bf16) staged via ASYNC global->LDS
// copies (rows padded to 72 halves = 144B: 36-dword stride is a bank
// permutation over the 16 fragment lanes). 8 waves = 8 stacked i-tiles.
// grid = (T/16 j-tiles, T/128 i-groups, B*H).
// ---------------------------------------------------------------------------
__global__ __launch_bounds__(256)
void mk_scores(const bf16_t* __restrict__ Q, const bf16_t* __restrict__ Km,
               const float* __restrict__ tco, const float* __restrict__ sco,
               float* __restrict__ attn, float scale)
{
    const int bh = blockIdx.z;
    const int b  = bh >> 4;          // H = 16
    const int h  = bh & 15;
    const int jBase  = blockIdx.x * 16;
    const int iBase0 = blockIdx.y * 128;
    const int tid  = threadIdx.x;
    const int lane = tid & 31;
    const int wave = tid >> 5;
    const int mRow = lane & 15;
    const int half = lane >> 4;

    __shared__ bf16_t Qs[128 * 72];  // [iLoc][d], 144B rows (16B aligned)
    __shared__ bf16_t Ks[16 * 72];   // [jLoc][d]
    __shared__ float tI[128], xI[128], yI[128];
    __shared__ float tJ[16],  xJ[16],  yJ[16];

    // --- async-stage Q tile: 128 rows x 128B = 1024 x 16B chunks ---
    #pragma unroll
    for (int it = 0; it < 4; ++it) {
        const int c   = tid + it * 256;
        const int row = c >> 3;
        const int off = (c & 7) * 8;                 // bf16 elements
        copy16_g2l(Q + ((size_t)b * kT + iBase0 + row) * kD + h * kHD + off,
                   &Qs[row * 72 + off]);
    }
    // --- async-stage K tile: 16 rows x 128B = 128 x 16B chunks ---
    if (tid < 128) {
        const int row = tid >> 3;
        const int off = (tid & 7) * 8;
        copy16_g2l(Km + ((size_t)b * kT + jBase + row) * kD + h * kHD + off,
                   &Ks[row * 72 + off]);
    }
    // --- stage coords ---
    if (tid < 128) {
        const int i = iBase0 + tid;
        tI[tid] = tco[b * kT + i];
        xI[tid] = sco[((size_t)b * kT + i) * 2 + 0];
        yI[tid] = sco[((size_t)b * kT + i) * 2 + 1];
    } else if (tid < 144) {
        const int j = tid - 128;
        tJ[j] = tco[b * kT + jBase + j];
        xJ[j] = sco[((size_t)b * kT + jBase + j) * 2 + 0];
        yJ[j] = sco[((size_t)b * kT + jBase + j) * 2 + 1];
    }
    wait_async_copies();
    __syncthreads();

    const unsigned int* qrow32 = (const unsigned int*)&Qs[(wave * 16 + mRow) * 72];
    const unsigned int* krow32 = (const unsigned int*)&Ks[mRow * 72];

    v8f acc = {};
    #pragma unroll
    for (int ks = 0; ks < 2; ++ks) {          // d = ks*32 .. ks*32+31
        Frag16 af, bf;
        #pragma unroll
        for (int v = 0; v < 4; ++v) af.u[v]     = qrow32[ks * 16 + v + 8 * half];
        #pragma unroll
        for (int v = 0; v < 4; ++v) af.u[4 + v] = qrow32[ks * 16 + 4 + v + 8 * half];
        #pragma unroll
        for (int v = 0; v < 8; ++v) bf.u[v]     = krow32[ks * 16 + 8 * half + v];
        acc = wmma_bf16(af, bf, acc);
    }

    // epilogue: Minkowski penalty + store raw scores into attn region
    const int   j  = jBase + mRow;
    const float tj = tJ[mRow], xj = xJ[mRow], yj = yJ[mRow];
    float* out = attn + (size_t)bh * kT * kT;
    #pragma unroll
    for (int r = 0; r < 8; ++r) {
        const int iLoc = wave * 16 + r + 8 * half;
        const float dt = tj - tI[iLoc];
        const float dx = xj - xI[iLoc];
        const float dy = yj - yI[iLoc];
        const float ds2 = dx * dx + dy * dy - dt * dt;
        const float pen = ds2 > 0.0f ? ds2 : 0.0f;
        out[(size_t)(iBase0 + iLoc) * kT + j] = acc[r] * scale - pen;
    }
}

// ---------------------------------------------------------------------------
// Row softmax over attn (row length T=2048), in place. 256 thr, 8 elems/thr.
// ---------------------------------------------------------------------------
__global__ __launch_bounds__(256)
void mk_softmax(float* __restrict__ attn)
{
    float* p = attn + (size_t)blockIdx.x * kT;
    const int tid = threadIdx.x;

    float v[8];
    #pragma unroll
    for (int e = 0; e < 8; ++e) v[e] = p[tid + e * 256];

    float mx = v[0];
    #pragma unroll
    for (int e = 1; e < 8; ++e) mx = fmaxf(mx, v[e]);
    #pragma unroll
    for (int off = 16; off > 0; off >>= 1)
        mx = fmaxf(mx, __shfl_xor(mx, off, 32));

    __shared__ float redM[8], redS[8];
    if ((tid & 31) == 0) redM[tid >> 5] = mx;
    __syncthreads();
    float gmax = redM[0];
    #pragma unroll
    for (int w = 1; w < 8; ++w) gmax = fmaxf(gmax, redM[w]);

    float s = 0.0f;
    #pragma unroll
    for (int e = 0; e < 8; ++e) { v[e] = __expf(v[e] - gmax); s += v[e]; }
    #pragma unroll
    for (int off = 16; off > 0; off >>= 1)
        s += __shfl_xor(s, off, 32);
    if ((tid & 31) == 0) redS[tid >> 5] = s;
    __syncthreads();
    float gsum = 0.0f;
    #pragma unroll
    for (int w = 0; w < 8; ++w) gsum += redS[w];

    const float inv = 1.0f / gsum;
    #pragma unroll
    for (int e = 0; e < 8; ++e) p[tid + e * 256] = v[e] * inv;
}

// ---------------------------------------------------------------------------
// ctx[b,i,h*64+d] = sum_j attn[b,h,i,j] * V[b,j,h*64+d]
// WG covers 128 i-rows x 64 d-cols for one (b,h); 8 waves, 4 tiles each.
// ---------------------------------------------------------------------------
__global__ __launch_bounds__(256)
void mk_attn_v(const float* __restrict__ attn, const bf16_t* __restrict__ V,
               float* __restrict__ ctx)
{
    const int bh = blockIdx.y;
    const int b  = bh >> 4;
    const int h  = bh & 15;
    const int iBase = blockIdx.x * 128;

    __shared__ bf16_t As[128 * 34];   // [iLoc][jLoc]
    __shared__ bf16_t Bs[64  * 34];   // [d][jLoc]   (transposed V tile)

    const int tid  = threadIdx.x;
    const int lane = tid & 31;
    const int wave = tid >> 5;
    const int mRow = lane & 15;
    const int half = lane >> 4;

    v8f acc[4] = {v8f{}, v8f{}, v8f{}, v8f{}};
    const float* arow = attn + ((size_t)bh * kT + iBase) * kT;

    for (int j0 = 0; j0 < kT; j0 += 32) {
        // stage attn tile 128x32 (f32 -> bf16): 16 elems / thread
        {
            const int r  = tid >> 1;
            const int cb = (tid & 1) * 16;
            const float* ap = arow + (size_t)r * kT + j0 + cb;
            unsigned int* dst = (unsigned int*)&As[r * 34 + cb];
            #pragma unroll
            for (int q = 0; q < 4; ++q) {
                float4 f = *(const float4*)(ap + q * 4);
                dst[q * 2 + 0] = pack2bf(f.x, f.y);
                dst[q * 2 + 1] = pack2bf(f.z, f.w);
            }
            if (j0 + 32 < kT) __builtin_prefetch(ap + 32, 0, 1);
        }
        // stage V tile 32(j) x 64(d), transposed
        {
            const int jr = tid >> 3;           // 0..31
            const int db = (tid & 7) * 8;      // 0..56
            const bf16_t* vp = V + ((size_t)b * kT + j0 + jr) * kD + h * kHD + db;
            #pragma unroll
            for (int q = 0; q < 8; ++q)
                Bs[(db + q) * 34 + jr] = vp[q];
        }
        __syncthreads();

        Frag16 af;
        {
            const unsigned int* ar = (const unsigned int*)&As[(wave * 16 + mRow) * 34];
            #pragma unroll
            for (int v = 0; v < 4; ++v) af.u[v]     = ar[v + 8 * half];
            #pragma unroll
            for (int v = 0; v < 4; ++v) af.u[4 + v] = ar[4 + v + 8 * half];
        }
        #pragma unroll
        for (int tn = 0; tn < 4; ++tn) {
            Frag16 bf;
            const unsigned int* bp = (const unsigned int*)&Bs[(tn * 16 + mRow) * 34 + 16 * half];
            #pragma unroll
            for (int v = 0; v < 8; ++v) bf.u[v] = bp[v];
            acc[tn] = wmma_bf16(af, bf, acc[tn]);
        }
        __syncthreads();
    }

    #pragma unroll
    for (int tn = 0; tn < 4; ++tn) {
        const int n = tn * 16 + mRow;
        #pragma unroll
        for (int r = 0; r < 8; ++r) {
            const int i = iBase + wave * 16 + r + 8 * half;
            ctx[((size_t)b * kT + i) * kD + h * kHD + n] = acc[tn][r];
        }
    }
}

// ---------------------------------------------------------------------------
extern "C" void kernel_launch(void* const* d_in, const int* in_sizes, int n_in,
                              void* d_out, int out_size, void* d_ws, size_t ws_size,
                              hipStream_t stream) {
    const float* x   = (const float*)d_in[0];
    const float* tco = (const float*)d_in[1];
    const float* sco = (const float*)d_in[2];
    const float* Wq  = (const float*)d_in[3];
    const float* bq  = (const float*)d_in[4];
    const float* Wk  = (const float*)d_in[5];
    const float* bk  = (const float*)d_in[6];
    const float* Wv  = (const float*)d_in[7];
    const float* bv  = (const float*)d_in[8];
    const float* Wo  = (const float*)d_in[9];
    const float* bo  = (const float*)d_in[10];

    const size_t MT = (size_t)kB * kT;          // 4096 rows
    char* ws = (char*)d_ws;
    bf16_t* Qb  = (bf16_t*)(ws);                         //  8 MB
    bf16_t* Kb  = (bf16_t*)(ws + MT * kD * 2);           //  8 MB
    bf16_t* Vb  = (bf16_t*)(ws + MT * kD * 4);           //  8 MB
    float*  ctx = (float*) (ws + MT * kD * 6);           // 16 MB

    float* out  = (float*)d_out;                         // (B,T,D)
    float* attn = out + MT * kD;                         // (B,H,T,T)

    const dim3 blk(256);
    const dim3 gProj(kD / 64, (int)(MT / 64));           // (16, 64)

    // Q/K/V projections (bf16 outputs)
    mk_gemm_bias<<<gProj, blk, 0, stream>>>(x, Wq, bq, nullptr, Qb, (int)MT, kD, kD);
    mk_gemm_bias<<<gProj, blk, 0, stream>>>(x, Wk, bk, nullptr, Kb, (int)MT, kD, kD);
    mk_gemm_bias<<<gProj, blk, 0, stream>>>(x, Wv, bv, nullptr, Vb, (int)MT, kD, kD);

    // scores with Minkowski penalty -> attn region
    const dim3 gS(kT / 16, kT / 128, kB * kH);           // (128, 16, 32)
    mk_scores<<<gS, blk, 0, stream>>>(Qb, Kb, tco, sco, attn, 0.125f);

    // row softmax in place
    mk_softmax<<<dim3(kB * kH * kT), blk, 0, stream>>>(attn);

    // attn @ V -> ctx
    const dim3 gC(kT / 128, kB * kH);                    // (16, 32)
    mk_attn_v<<<gC, blk, 0, stream>>>(attn, Vb, ctx);

    // output projection (f32 output)
    mk_gemm_bias<<<gProj, blk, 0, stream>>>(ctx, Wo, bo, out, nullptr, (int)MT, kD, kD);
}
